// NonLocalAttention_11450382811413
// MI455X (gfx1250) — matrix operand order repaired and checked by reading, not compiled, not run on hardware
//
#include <hip/hip_runtime.h>

// ---------------------------------------------------------------------------
// NonLocalAttention for MI455X (gfx1250): bf16 WMMA flash attention with
// TDM (tensor_load_to_lds) double-buffered K/V staging.
//   B=8, C=192, Cr=24 (padded to 32), HW=4096.
// ---------------------------------------------------------------------------

typedef __attribute__((ext_vector_type(16))) __bf16 v16bf;
typedef __attribute__((ext_vector_type(8)))  float  v8f;
typedef __attribute__((ext_vector_type(4)))  unsigned int u32x4v;
typedef __attribute__((ext_vector_type(8)))  int          i32x8v;
typedef __attribute__((ext_vector_type(4)))  int          i32x4v;

union FragBF {
    v16bf v;
    uint4 q[2];
    unsigned short us[16];
};

__device__ inline unsigned short f2bf(float f) {
    unsigned int u = __float_as_uint(f);
    unsigned int r = u + 0x7FFFu + ((u >> 16) & 1u);
    return (unsigned short)(r >> 16);
}

__device__ inline v8f wmma_bf16(v16bf a, v16bf b, v8f c) {
    return __builtin_amdgcn_wmma_f32_16x16x32_bf16(
        /*neg_a=*/false, a, /*neg_b=*/false, b,
        /*c_mod=*/(short)0, c, /*reuse_a=*/false, /*reuse_b=*/false);
}

// --- Tensor Data Mover: 2D bf16 tile -> LDS with 16B padding per 64B row ---
// D# per CDNA5 ISA 8.3/8.4. pad_interval=3 (16 DWORDs), pad_amount=3 (4 DWORDs)
// => LDS row stride = 64B data + 16B pad = 80B (40 bf16), matching Ksh/Vsh.
// This toolchain's builtin is the 6-arg form:
//   (u32x4 group0, i32x8 group1, i32x4 group2, i32x4 group3, i32x8 pad, i32 cpol)
__device__ inline void tdm_load_2d_bf16(unsigned lds_byte_addr, const void* gptr,
                                        unsigned tensor_d0, unsigned tensor_d1,
                                        unsigned tile_d0, unsigned tile_d1,
                                        unsigned long long stride0) {
    unsigned long long ga = (unsigned long long)(uintptr_t)gptr;
    u32x4v g0;
    g0[0] = 1u;                                                  // count=1, user mode
    g0[1] = lds_byte_addr;                                       // lds_addr
    g0[2] = (unsigned)(ga & 0xFFFFFFFFu);                        // global_addr lo
    g0[3] = (unsigned)((ga >> 32) & 0x1FFFFFFu) | (2u << 30);    // addr hi | type=2
    i32x8v g1;
    g1[0] = (int)((1u << 16) | (1u << 20) | (3u << 22) | (3u << 25)); // 2B, pad en
    g1[1] = (int)((tensor_d0 & 0xFFFFu) << 16);                  // abar=0 | td0[15:0]
    g1[2] = (int)((tensor_d0 >> 16) | ((tensor_d1 & 0xFFFFu) << 16));
    g1[3] = (int)((tensor_d1 >> 16) | (tile_d0 << 16));
    g1[4] = (int)(tile_d1 & 0xFFFFu);                            // tile_dim2=0
    g1[5] = (int)(stride0 & 0xFFFFFFFFu);
    g1[6] = (int)((stride0 >> 32) & 0xFFFFu);                    // dim1_stride=0
    g1[7] = 0;
    i32x4v z4 = {0, 0, 0, 0};
    i32x8v z8 = {0, 0, 0, 0, 0, 0, 0, 0};
    __builtin_amdgcn_tensor_load_to_lds(g0, g1, z4, z4, z8, 0);
}

__device__ inline unsigned lds_off(const void* p) {
    // Generic LDS pointer: flat addr low 32 bits = LDS byte offset (ISA 10.2).
    return (unsigned)(uintptr_t)p;
}

#define NB   8
#define NC   192
#define NCR  24
#define NHW  4096
#define NROWS 240   // 24 (q) + 24 (k) + 192 (v)

// ---- workspace layout (bytes) ----
#define QH_OFF    ((size_t)0)                       // [B][HW][32] bf16, ch 24..31 zero
#define KH_OFF    ((size_t)(NB * NHW * 32 * 2))     // 2,097,152
#define VH_OFF    (KH_OFF + (size_t)(NB * NHW * 32 * 2))          // [B][192][HW] bf16
#define WBF_OFF   (VH_OFF + (size_t)(NB * NC * NHW * 2))          // [240][192] bf16
#define BIAS_OFF  (WBF_OFF + (size_t)(NROWS * NC * 2))            // [240] f32
#define SLOPE_OFF (BIAS_OFF + (size_t)(NROWS * 4))                // [240] f32

// ===========================================================================
// Kernel 0: pack weights to bf16, gather bias/slope, zero padded Q/K buffers.
// ===========================================================================
__global__ __launch_bounds__(256) void nla_prep(
    const float* __restrict__ w1, const float* __restrict__ b1, const float* __restrict__ a1,
    const float* __restrict__ w2, const float* __restrict__ b2, const float* __restrict__ a2,
    const float* __restrict__ wa, const float* __restrict__ ba, const float* __restrict__ aa,
    unsigned int* __restrict__ qkzero,
    unsigned short* __restrict__ wbf, float* __restrict__ bias, float* __restrict__ slope)
{
    int g = blockIdx.x * 256 + threadIdx.x;
    const int NZ = (NB * NHW * 32 * 2 * 2) / 4;   // u32 words covering Qh+Kh
    if (g < NZ) qkzero[g] = 0u;
    if (g < NROWS * NC) {
        int r = g / NC, c = g % NC;
        float v = (r < NCR)     ? w1[r * NC + c]
                : (r < 2 * NCR) ? w2[(r - NCR) * NC + c]
                                : wa[(r - 2 * NCR) * NC + c];
        wbf[g] = f2bf(v);
    }
    if (g < NROWS) {
        bias[g]  = (g < NCR) ? b1[g] : (g < 2 * NCR) ? b2[g - NCR] : ba[g - 2 * NCR];
        slope[g] = (g < NCR) ? a1[0] : (g < 2 * NCR) ? a2[0]       : aa[0];
    }
}

// ===========================================================================
// Kernel 1: projection GEMM  P[240, HW] = prelu(W[240,192] @ X[192,HW] + b)
// ===========================================================================
__global__ __launch_bounds__(256) void nla_proj(
    const float* __restrict__ x,               // [B][192][HW] f32
    const unsigned short* __restrict__ wbf,    // [240][192] bf16
    const float* __restrict__ bias, const float* __restrict__ slope,
    unsigned short* __restrict__ qh,           // [B][HW][32] bf16
    unsigned short* __restrict__ kh,           // [B][HW][32] bf16
    unsigned short* __restrict__ vh)           // [B][192][HW] bf16
{
    constexpr int XS = 208;                    // 192 + 16 pad, 416B rows (16B aligned)
    __shared__ unsigned short xs[64 * XS];     // pixel-major [64 px][208 ch]

    const int b   = blockIdx.y;
    const int px0 = blockIdx.x * 64;
    const int tid = threadIdx.x;

    #pragma unroll 4
    for (int i = 0; i < 48; ++i) {
        int id = tid + i * 256;
        int ch = id >> 6, px = id & 63;
        float v = x[((size_t)b * NC + ch) * NHW + px0 + px];
        xs[px * XS + ch] = f2bf(v);
    }
    __syncthreads();

    const int wave  = tid >> 5;
    const int lane  = tid & 31;
    const int lhalf = lane & 15;
    const bool hi   = lane >= 16;

    for (int t = wave; t < 60; t += 8) {
        int rt = t >> 2, ct = t & 3;
        v8f acc = {};
        #pragma unroll
        for (int ks = 0; ks < 6; ++ks) {
            FragBF a, xb;
            const uint4* wp = (const uint4*)(wbf + (size_t)(rt * 16 + lhalf) * NC
                                                  + ks * 32 + (hi ? 8 : 0));
            a.q[0] = wp[0];
            a.q[1] = wp[2];
            const uint4* xp = (const uint4*)(xs + (ct * 16 + lhalf) * XS
                                                + ks * 32 + (hi ? 8 : 0));
            xb.q[0] = xp[0];
            xb.q[1] = xp[2];
            acc = wmma_bf16(a.v, xb.v, acc);
        }
        // Branchless PReLU epilogue with vector bias/slope loads.
        const int rbase = rt * 16 + (hi ? 8 : 0);
        const int px    = px0 + ct * 16 + lhalf;
        float4 bi0 = *(const float4*)(bias + rbase);
        float4 bi1 = *(const float4*)(bias + rbase + 4);
        float4 sl0 = *(const float4*)(slope + rbase);
        float4 sl1 = *(const float4*)(slope + rbase + 4);
        float bb[8] = {bi0.x, bi0.y, bi0.z, bi0.w, bi1.x, bi1.y, bi1.z, bi1.w};
        float ss[8] = {sl0.x, sl0.y, sl0.z, sl0.w, sl1.x, sl1.y, sl1.z, sl1.w};
        #pragma unroll
        for (int j = 0; j < 8; ++j) {
            int row = rbase + j;
            float v = acc[j] + bb[j];
            v = fmaxf(v, 0.f) + ss[j] * fminf(v, 0.f);
            unsigned short bv = f2bf(v);
            if (row < NCR)
                qh[((size_t)b * NHW + px) * 32 + row] = bv;
            else if (row < 2 * NCR)
                kh[((size_t)b * NHW + px) * 32 + (row - NCR)] = bv;
            else
                vh[((size_t)b * NC + (row - 2 * NCR)) * NHW + px] = bv;
        }
    }
}

// ===========================================================================
// Kernel 2: flash attention, transposed dataflow (queries on lanes).
//   S^T[32k x 16q] = K·Q^T   (2 WMMA)
//   per-lane online softmax  (stats are lane-scalars, one xor-16 shuffle each)
//   O^T[192c x 16q] += V^T·P^T (12 WMMA)
// K/V tiles DMA'd by the Tensor Data Mover (double buffered, TENSORcnt).
// ===========================================================================
#define KSH_ELEMS (32 * 40)     // [32 keys][32 ch + 8 pad]  (pad from TDM)
#define VSH_ELEMS (192 * 40)    // [192 ch][32 keys + 8 pad]
#define KSH_B     ((size_t)0)
#define VSH_B     ((size_t)(2 * KSH_ELEMS * 2))                 // 5120
#define SMEM_BYTES 55296        // Osh [192][72] f32 overlaps K/V after the loop

__global__ __launch_bounds__(128) void nla_attn(
    const unsigned short* __restrict__ qh,     // [B][HW][32]
    const unsigned short* __restrict__ kh,     // [B][HW][32]
    const unsigned short* __restrict__ vh,     // [B][192][HW]
    float* __restrict__ out)                   // [B][192][HW]
{
    extern __shared__ __align__(16) char smem[];
    unsigned short* Ksh = (unsigned short*)(smem + KSH_B);
    unsigned short* Vsh = (unsigned short*)(smem + VSH_B);
    float*          Osh = (float*)smem;        // [192][72] (post-loop)

    const int b     = blockIdx.y;
    const int q0    = blockIdx.x * 64;
    const int tid   = threadIdx.x;
    const int wave  = tid >> 5;
    const int lane  = tid & 31;
    const int lhalf = lane & 15;
    const bool hi   = lane >= 16;

    // Q^T B-fragment (column = query), loaded once, reused 128 iterations.
    FragBF qb;
    {
        const uint4* qp = (const uint4*)(qh + ((size_t)b * NHW + q0 + wave * 16 + lhalf) * 32
                                            + (hi ? 8 : 0));
        qb.q[0] = qp[0];
        qb.q[1] = qp[2];
    }

    v8f O[12];
    #pragma unroll
    for (int t = 0; t < 12; ++t) O[t] = (v8f){};
    float m = -3.0e38f, l = 0.f;

    auto issue_tiles = [&](int kt, int buf) {
        // K tile: [32 keys][32 ch] from kh row kt*32
        tdm_load_2d_bf16(lds_off(Ksh + buf * KSH_ELEMS),
                         kh + ((size_t)b * NHW + kt * 32) * 32,
                         /*td0=*/32, /*td1=*/NHW, /*tile0=*/32, /*tile1=*/32,
                         /*stride0=*/32ull);
        // V tile: [192 ch][32 keys] from vh column kt*32
        tdm_load_2d_bf16(lds_off(Vsh + buf * VSH_ELEMS),
                         vh + (size_t)b * NC * NHW + kt * 32,
                         /*td0=*/NHW, /*td1=*/NC, /*tile0=*/32, /*tile1=*/NC,
                         /*stride0=*/(unsigned long long)NHW);
    };

    if (wave == 0) {
        issue_tiles(0, 0);
        __builtin_amdgcn_s_wait_tensorcnt(0);
    }
    __syncthreads();

    for (int kt = 0; kt < 128; ++kt) {
        const int buf = kt & 1;
        if (wave == 0 && kt + 1 < 128) issue_tiles(kt + 1, buf ^ 1);  // async DMA

        // ---- S^T = K @ Q^T : A = K rows (keys x ch), B = Q^T ----
        const unsigned short* kb = Ksh + buf * KSH_ELEMS;
        FragBF ka0, ka1;
        {
            const uint4* p0 = (const uint4*)(kb + (lhalf)      * 40 + (hi ? 8 : 0));
            const uint4* p1 = (const uint4*)(kb + (16 + lhalf) * 40 + (hi ? 8 : 0));
            ka0.q[0] = p0[0]; ka0.q[1] = p0[2];
            ka1.q[0] = p1[0]; ka1.q[1] = p1[2];
        }
        v8f s0 = {}, s1 = {};
        s0 = wmma_bf16(ka0.v, qb.v, s0);   // keys 0..15  (M) x queries (N=lanes)
        s1 = wmma_bf16(ka1.v, qb.v, s1);   // keys 16..31

        // ---- online softmax: per-lane (query) stats ----
        float r = fmaxf(s0[0], s1[0]);
        #pragma unroll
        for (int j = 1; j < 8; ++j) r = fmaxf(r, fmaxf(s0[j], s1[j]));
        r = fmaxf(r, __shfl_xor(r, 16, 32));     // combine key-halves
        float mn   = fmaxf(m, r);
        float corr = __expf(m - mn);
        m = mn;
        float p0v[8], p1v[8];
        float rs = 0.f;
        #pragma unroll
        for (int j = 0; j < 8; ++j) {
            p0v[j] = __expf(s0[j] - mn);
            p1v[j] = __expf(s1[j] - mn);
            rs += p0v[j] + p1v[j];
        }
        rs += __shfl_xor(rs, 16, 32);
        l = l * corr + rs;
        #pragma unroll
        for (int t = 0; t < 12; ++t)
            #pragma unroll
            for (int j = 0; j < 8; ++j) O[t][j] *= corr;

        // ---- build P^T B-fragment in registers (cross-half swap + pack) ----
        // lo lane n: keys 0..7 own (s0), keys 8..15 from partner lane n+16 (s0)
        // hi lane n: keys 16..23 from partner lane n-16 (s1), keys 24..31 own (s1)
        FragBF pb;
        #pragma unroll
        for (int j = 0; j < 8; ++j) {
            float sw0 = __shfl_xor(p0v[j], 16, 32);
            float sw1 = __shfl_xor(p1v[j], 16, 32);
            float lo8 = hi ? sw1 : p0v[j];
            float hi8 = hi ? p1v[j] : sw0;
            pb.us[j]     = f2bf(lo8);
            pb.us[8 + j] = f2bf(hi8);
        }

        // ---- O^T += V^T @ P^T : A = V^T (ch x keys), B = P^T ----
        const unsigned short* vb = Vsh + buf * VSH_ELEMS;
        #pragma unroll
        for (int t = 0; t < 12; ++t) {
            FragBF va;
            const uint4* p = (const uint4*)(vb + (t * 16 + lhalf) * 40 + (hi ? 16 : 0));
            va.q[0] = p[0];
            va.q[1] = p[1];
            O[t] = wmma_bf16(va.v, pb.v, O[t]);
        }

        if (wave == 0) __builtin_amdgcn_s_wait_tensorcnt(0);
        __syncthreads();                      // single barrier per iteration
    }

    // ---- epilogue: normalize, stage [192ch][64px] in LDS, coalesced store ----
    float inv = 1.0f / l;
    #pragma unroll
    for (int t = 0; t < 12; ++t) {
        int chb = t * 16 + (hi ? 8 : 0);
        int qq  = wave * 16 + lhalf;
        #pragma unroll
        for (int j = 0; j < 8; ++j)
            Osh[(chb + j) * 72 + qq] = O[t][j] * inv;   // 16-lane contiguous rows
    }
    __syncthreads();
    #pragma unroll 4
    for (int i = 0; i < 24; ++i) {
        int id = tid + i * 128;                // 0..3071 float4 chunks
        int row = id >> 4, chunk = id & 15;
        float4 vv = *(const float4*)(Osh + row * 72 + chunk * 4);
        *(float4*)(out + ((size_t)b * NC + row) * NHW + q0 + chunk * 4) = vv;
    }
}

// ===========================================================================
extern "C" void kernel_launch(void* const* d_in, const int* in_sizes, int n_in,
                              void* d_out, int out_size, void* d_ws, size_t ws_size,
                              hipStream_t stream) {
    const float* x  = (const float*)d_in[0];
    const float* w1 = (const float*)d_in[1];
    const float* b1 = (const float*)d_in[2];
    const float* a1 = (const float*)d_in[3];
    const float* w2 = (const float*)d_in[4];
    const float* b2 = (const float*)d_in[5];
    const float* a2 = (const float*)d_in[6];
    const float* wa = (const float*)d_in[7];
    const float* ba = (const float*)d_in[8];
    const float* aa = (const float*)d_in[9];
    float* out = (float*)d_out;

    char* ws = (char*)d_ws;
    unsigned short* qh    = (unsigned short*)(ws + QH_OFF);
    unsigned short* kh    = (unsigned short*)(ws + KH_OFF);
    unsigned short* vh    = (unsigned short*)(ws + VH_OFF);
    unsigned short* wbf   = (unsigned short*)(ws + WBF_OFF);
    float*          bias  = (float*)(ws + BIAS_OFF);
    float*          slope = (float*)(ws + SLOPE_OFF);

    nla_prep<<<4096, 256, 0, stream>>>(w1, b1, a1, w2, b2, a2, wa, ba, aa,
                                       (unsigned int*)(ws + QH_OFF), wbf, bias, slope);
    nla_proj<<<dim3(NHW / 64, NB), 256, 0, stream>>>(x, wbf, bias, slope, qh, kh, vh);
    nla_attn<<<dim3(NHW / 64, NB), 128, SMEM_BYTES, stream>>>(qh, kh, vh, out);
}
